// EncoderLayer_35261681500409
// MI455X (gfx1250) — matrix operand order repaired and checked by reading, not compiled, not run on hardware
//
#include <hip/hip_runtime.h>
#include <hip/hip_bf16.h>
#include <math.h>

// Problem sizes (fixed by the reference)
#define BB   4
#define LL   2048
#define DD   1024
#define HH   16
#define FF   4096
#define HD   64          // head dim
#define BHN  64          // B*H
#define NT   8192        // B*L
#define USN  2048        // U_part = len(index_sample)
#define UU   40          // u (top-k queries)
#define UP   48          // u padded to multiple of 16

typedef __attribute__((ext_vector_type(16))) __bf16 v16bf;
typedef __attribute__((ext_vector_type(8)))  float  v8f;

union FragAB { v16bf v; uint4 q[2]; unsigned short h[16]; };

__device__ __forceinline__ unsigned short f32_bf16(float f) {
  unsigned int u = __float_as_uint(f);
  u += 0x7FFFu + ((u >> 16) & 1u);           // round-to-nearest-even
  return (unsigned short)(u >> 16);
}

// monotonic-int trick: works for mixed-sign floats, integer atomics only
__device__ __forceinline__ void atomic_max_f32(float* addr, float val) {
  if (val >= 0.0f) atomicMax((int*)addr, __float_as_int(val));
  else             atomicMin((unsigned int*)addr, __float_as_uint(val));
}

// ---------------------------------------------------------------------------
// CDNA5 async global->LDS copies (GLOBAL_LOAD_ASYNC_TO_LDS_B128, ASYNCcnt)
// Builtin signature (from clang diagnostic): (global int4*, local int4*, Ii, Ii)
// ---------------------------------------------------------------------------
#if defined(__has_builtin)
#if __has_builtin(__builtin_amdgcn_global_load_async_to_lds_b128)
#define HAVE_ASYNC_LDS 1
#endif
#endif

#if defined(HAVE_ASYNC_LDS)
typedef int v4i32 __attribute__((vector_size(16)));
typedef __attribute__((address_space(1))) v4i32* gas_v4i32_ptr;
typedef __attribute__((address_space(3))) v4i32* las_v4i32_ptr;
__device__ __forceinline__ void async_ld16(const unsigned short* g,
                                           unsigned short* l) {
  __builtin_amdgcn_global_load_async_to_lds_b128(
      (gas_v4i32_ptr)g, (las_v4i32_ptr)l, 0, 0);
}
#if __has_builtin(__builtin_amdgcn_s_wait_asynccnt)
#define WAIT_ASYNC(n) __builtin_amdgcn_s_wait_asynccnt(n)
#else
#define WAIT_ASYNC(n) asm volatile("s_wait_asynccnt " #n ::: "memory")
#endif
#endif

// ---------------------------------------------------------------------------
// Elementwise converts
// ---------------------------------------------------------------------------
__global__ void cvt_bf16_kernel(const float* __restrict__ s,
                                unsigned short* __restrict__ d, int n) {
  int i = blockIdx.x * 256 + threadIdx.x;
  if (i < n) d[i] = f32_bf16(s[i]);
}

// dst(C,R) = transpose(src(R,C)), f32 -> bf16
__global__ __launch_bounds__(256)
void cvt_bf16_t_kernel(const float* __restrict__ s,
                       unsigned short* __restrict__ d, int R, int C) {
  __shared__ float tile[32][33];
  int cb = blockIdx.x * 32, rb = blockIdx.y * 32;
  int tx = threadIdx.x & 31, ty = threadIdx.x >> 5;
#pragma unroll
  for (int i = 0; i < 32; i += 8)
    tile[ty + i][tx] = s[(size_t)(rb + ty + i) * C + cb + tx];
  __syncthreads();
#pragma unroll
  for (int i = 0; i < 32; i += 8)
    d[(size_t)(cb + ty + i) * R + rb + tx] = f32_bf16(tile[tx][ty + i]);
}

__global__ void zero_u16_kernel(unsigned short* p, int n) {
  int i = blockIdx.x * 256 + threadIdx.x;
  if (i < n) p[i] = 0;
}

// ---------------------------------------------------------------------------
// Big tiled WMMA GEMM:  C(M,N) = A(M,K) * BT(N,K)^T + bias
// block tile 128x128, 8 waves, wave tile 32x64 (2x4 WMMA 16x16x32 tiles)
// Async path: double-buffered LDS, GLOBAL_LOAD_ASYNC_TO_LDS_B128 pipeline
// with s_wait_asynccnt draining only the current tile's copies.
// All dims assumed multiples of the tile sizes (true for this problem).
// ---------------------------------------------------------------------------
#define GM_BM 128
#define GM_BN 128
#define GM_BK 32
#define LDSTR 40   // LDS row stride in halves (80B, 16B aligned, pads banks)

#define MODE_HEADS 0   // bf16 out, (B,H,L,d) head-major (used for k)
#define MODE_VT    1   // bf16 out, (B,H,d,L) head-major transposed (v)
#define MODE_F32   2   // f32  out, row-major (wo, ffn2)
#define MODE_RELU  3   // bf16 out, row-major, relu (ffn1)
#define MODE_PLAIN 4   // bf16 out, row-major (q: flat reshape, NO transpose)

__global__ __launch_bounds__(256)
void gemm_bf16_kernel(const unsigned short* __restrict__ A, int lda,
                      const unsigned short* __restrict__ BT, int ldb,
                      const float* __restrict__ bias,
                      int N, int K, int mode, void* __restrict__ outp) {
#if defined(HAVE_ASYNC_LDS)
  __shared__ unsigned short sA[2][GM_BM * LDSTR];
  __shared__ unsigned short sB[2][GM_BN * LDSTR];
#else
  __shared__ unsigned short sA[1][GM_BM * LDSTR];
  __shared__ unsigned short sB[1][GM_BN * LDSTR];
#endif
  const int tid  = threadIdx.x;
  const int lane = tid & 31, wave = tid >> 5;
  const int wm = wave >> 1, wn = wave & 1;
  const int r16 = lane & 15, hi = lane >> 4;
  const int rowBase = blockIdx.y * GM_BM;
  const int colBase = blockIdx.x * GM_BN;

  v8f acc[2][4] = {};

  // per-thread staging coordinates: thread t covers LDS row t/2, 2 segments
  const int sr = tid >> 1;               // LDS row (0..127)
  const int ss = (tid & 1) * 2;          // first 8-half segment (0 or 2)

  auto compute_tile = [&](int buf) {
    FragAB af[2], bfr[4];
#pragma unroll
    for (int i = 0; i < 2; ++i) {   // A frag: ISA 16-bit A 16x32 layout
      const unsigned short* p = &sA[buf][(wm * 32 + i * 16 + r16) * LDSTR];
      af[i].q[0] = *(const uint4*)(p + hi * 8);
      af[i].q[1] = *(const uint4*)(p + 16 + hi * 8);
    }
#pragma unroll
    for (int j = 0; j < 4; ++j) {   // B frag from transposed tile (n,k)
      const unsigned short* p = &sB[buf][(wn * 64 + j * 16 + r16) * LDSTR + hi * 16];
      bfr[j].q[0] = *(const uint4*)(p - hi * 8);
      bfr[j].q[1] = *(const uint4*)(p + 16 - hi * 8);
    }
#pragma unroll
    for (int i = 0; i < 2; ++i)
#pragma unroll
      for (int j = 0; j < 4; ++j)
        acc[i][j] = __builtin_amdgcn_wmma_f32_16x16x32_bf16(
            false, af[i].v, false, bfr[j].v, (short)0, acc[i][j], false, false);
  };

#if defined(HAVE_ASYNC_LDS)
  auto stage_async = [&](int kk, int buf) {
#pragma unroll
    for (int s2 = 0; s2 < 2; ++s2) {
      int seg = ss + s2;
      async_ld16(A  + (size_t)(rowBase + sr) * lda + kk + seg * 8,
                 &sA[buf][sr * LDSTR + seg * 8]);
      async_ld16(BT + (size_t)(colBase + sr) * ldb + kk + seg * 8,
                 &sB[buf][sr * LDSTR + seg * 8]);
    }
  };
  const int nk = K / GM_BK;
  stage_async(0, 0);
  for (int t = 0; t < nk; ++t) {
    const int buf = t & 1;
    if (t + 1 < nk) {
      stage_async((t + 1) * GM_BK, buf ^ 1);  // stream next tile
      WAIT_ASYNC(4);                          // drain current tile only
    } else {
      WAIT_ASYNC(0);
    }
    __syncthreads();
    compute_tile(buf);
    __syncthreads();                          // readers done before re-stage
  }
#else
  for (int kk = 0; kk < K; kk += GM_BK) {
    // batched synchronous stage: issue all loads, then all stores
    uint4 ta0 = *(const uint4*)(A  + (size_t)(rowBase + sr) * lda + kk + ss * 8);
    uint4 ta1 = *(const uint4*)(A  + (size_t)(rowBase + sr) * lda + kk + (ss + 1) * 8);
    uint4 tb0 = *(const uint4*)(BT + (size_t)(colBase + sr) * ldb + kk + ss * 8);
    uint4 tb1 = *(const uint4*)(BT + (size_t)(colBase + sr) * ldb + kk + (ss + 1) * 8);
    *(uint4*)(&sA[0][sr * LDSTR + ss * 8])       = ta0;
    *(uint4*)(&sA[0][sr * LDSTR + (ss + 1) * 8]) = ta1;
    *(uint4*)(&sB[0][sr * LDSTR + ss * 8])       = tb0;
    *(uint4*)(&sB[0][sr * LDSTR + (ss + 1) * 8]) = tb1;
    if (kk + GM_BK < K) {   // emits global_prefetch_b8 for next K tiles
      __builtin_prefetch(A  + (size_t)(rowBase + sr) * lda + kk + GM_BK, 0, 3);
      __builtin_prefetch(BT + (size_t)(colBase + sr) * ldb + kk + GM_BK, 0, 3);
    }
    __syncthreads();
    compute_tile(0);
    __syncthreads();
  }
#endif

  // Epilogue.  C layout: VGPR r -> row = r + 8*hi, col = lane&15
#pragma unroll
  for (int i = 0; i < 2; ++i) {
#pragma unroll
    for (int j = 0; j < 4; ++j) {
      int col = colBase + wn * 64 + j * 16 + r16;
      float bvv = bias ? bias[col] : 0.0f;
      int row0 = rowBase + wm * 32 + i * 16 + hi * 8;
#pragma unroll
      for (int r = 0; r < 8; ++r) {
        int row = row0 + r;
        float val = acc[i][j][r] + bvv;
        if (mode == MODE_HEADS) {
          int b = row >> 11, l = row & (LL - 1);
          int h = col >> 6,  dd2 = col & (HD - 1);
          ((unsigned short*)outp)[(((size_t)(b * HH + h)) * LL + l) * HD + dd2] =
              f32_bf16(val);
        } else if (mode == MODE_VT) {
          int b = row >> 11, l = row & (LL - 1);
          int h = col >> 6,  dd2 = col & (HD - 1);
          ((unsigned short*)outp)[(((size_t)(b * HH + h)) * HD + dd2) * LL + l] =
              f32_bf16(val);
        } else if (mode == MODE_F32) {
          ((float*)outp)[(size_t)row * N + col] = val;
        } else if (mode == MODE_RELU) {
          ((unsigned short*)outp)[(size_t)row * N + col] =
              f32_bf16(fmaxf(val, 0.0f));
        } else { // MODE_PLAIN
          ((unsigned short*)outp)[(size_t)row * N + col] = f32_bf16(val);
        }
      }
    }
  }
}

// ---------------------------------------------------------------------------
// Gather K_sample rows: ksamp[bh][j][:] = k[bh][index_sample[j]][:]
// ---------------------------------------------------------------------------
__global__ void gather_ksamp_kernel(const unsigned short* __restrict__ kh,
                                    const int* __restrict__ idxs,
                                    unsigned short* __restrict__ ksamp) {
  int i = blockIdx.x * 256 + threadIdx.x;     // BHN*USN*HD total
  int dd2 = i & (HD - 1);
  int j   = (i >> 6) & (USN - 1);
  int bh  = i >> 17;                          // USN*HD = 2^17
  ksamp[i] = kh[((size_t)bh * LL + idxs[j]) * HD + dd2];
}

__global__ void init_m_kernel(float* Mmax, float* Msum) {
  int i = blockIdx.x * 256 + threadIdx.x;
  if (i < BHN * LL) { Mmax[i] = -INFINITY; Msum[i] = 0.0f; }
}

// ---------------------------------------------------------------------------
// QK sample pass with fused M-statistics.  Per bh: scores = q(L,64)@ksamp^T.
// Wave tile 32x32 (2x2 WMMA), fragments loaded straight from global (K=64).
// Row max/sum reduced across the 16-col groups via shuffles, then one
// integer-atomic max + one atomicAdd per (row, 32-col strip).
// ---------------------------------------------------------------------------
__global__ __launch_bounds__(256)
void qk_m_kernel(const unsigned short* __restrict__ qh,
                 const unsigned short* __restrict__ ksamp,
                 float* __restrict__ Mmax, float* __restrict__ Msum) {
  const int bh = blockIdx.z;
  const int tid = threadIdx.x;
  const int lane = tid & 31, wave = tid >> 5;
  const int wm = wave >> 1, wn = wave & 1;
  const int r16 = lane & 15, hi = lane >> 4;
  const int rowBase = blockIdx.y * 128 + wm * 32;
  const int colBase = blockIdx.x * 64 + wn * 32;
  const unsigned short* Ab = qh    + (size_t)bh * LL * HD;
  const unsigned short* Bb = ksamp + (size_t)bh * USN * HD;

  v8f acc[2][2] = {};
#pragma unroll
  for (int ks = 0; ks < HD; ks += 32) {
    FragAB af[2], bfr[2];
#pragma unroll
    for (int i = 0; i < 2; ++i) {
      const unsigned short* p = Ab + (size_t)(rowBase + i * 16 + r16) * HD + ks;
      af[i].q[0] = *(const uint4*)(p + hi * 8);
      af[i].q[1] = *(const uint4*)(p + 16 + hi * 8);
    }
#pragma unroll
    for (int j = 0; j < 2; ++j) {
      const unsigned short* p = Bb + (size_t)(colBase + j * 16 + r16) * HD + ks + hi * 16;
      bfr[j].q[0] = *(const uint4*)(p - hi * 8);
      bfr[j].q[1] = *(const uint4*)(p + 16 - hi * 8);
    }
#pragma unroll
    for (int i = 0; i < 2; ++i)
#pragma unroll
      for (int j = 0; j < 2; ++j)
        acc[i][j] = __builtin_amdgcn_wmma_f32_16x16x32_bf16(
            false, af[i].v, false, bfr[j].v, (short)0, acc[i][j], false, false);
  }

#pragma unroll
  for (int i = 0; i < 2; ++i) {
#pragma unroll
    for (int r = 0; r < 8; ++r) {
      float mx = fmaxf(acc[i][0][r], acc[i][1][r]);
      float sm = acc[i][0][r] + acc[i][1][r];
#pragma unroll
      for (int m = 8; m >= 1; m >>= 1) {   // xor<=8 stays within 16-lane half
        mx = fmaxf(mx, __shfl_xor(mx, m, 32));
        sm += __shfl_xor(sm, m, 32);
      }
      if (r16 == 0) {
        int row = rowBase + i * 16 + hi * 8 + r;
        atomic_max_f32(&Mmax[(size_t)bh * LL + row], mx);
        atomicAdd(&Msum[(size_t)bh * LL + row], sm);
      }
    }
  }
}

// ---------------------------------------------------------------------------
// M = max - sum/L; select UU smallest (reference does top_k(-M)).
// One workgroup per bh; iterative argmin over LDS copy.
// ---------------------------------------------------------------------------
__global__ __launch_bounds__(256)
void topk_kernel(const float* __restrict__ Mmax, const float* __restrict__ Msum,
                 int* __restrict__ Mtop) {
  __shared__ float sM[LL];
  __shared__ float rv[256];
  __shared__ int   ri[256];
  int bh = blockIdx.x, tid = threadIdx.x;
  for (int i = tid; i < LL; i += 256)
    sM[i] = Mmax[(size_t)bh * LL + i] - Msum[(size_t)bh * LL + i] * (1.0f / LL);
  __syncthreads();
  for (int t = 0; t < UU; ++t) {
    float best = INFINITY; int bi = -1;
    for (int i = tid; i < LL; i += 256)
      if (sM[i] < best) { best = sM[i]; bi = i; }
    rv[tid] = best; ri[tid] = bi;
    __syncthreads();
    for (int s = 128; s > 0; s >>= 1) {
      if (tid < s && rv[tid + s] < rv[tid]) { rv[tid] = rv[tid + s]; ri[tid] = ri[tid + s]; }
      __syncthreads();
    }
    if (tid == 0) { Mtop[bh * UU + t] = ri[0]; sM[ri[0]] = INFINITY; }
    __syncthreads();
  }
}

// qred[bh][t][:] = q[bh][Mtop[bh][t]][:]; rows UU..UP-1 zeroed
__global__ void gather_qred_kernel(const unsigned short* __restrict__ qh,
                                   const int* __restrict__ Mtop,
                                   unsigned short* __restrict__ qred) {
  int i = blockIdx.x * 256 + threadIdx.x;     // BHN*UP*HD
  int dd2 = i & (HD - 1);
  int qq  = i >> 6;
  int t = qq % UP, bh = qq / UP;
  unsigned short v = 0;
  if (t < UU) v = qh[((size_t)bh * LL + Mtop[bh * UU + t]) * HD + dd2];
  qred[i] = v;
}

// ---------------------------------------------------------------------------
// scores(bh, UP, L) = qred(UP,64) @ k(L,64)^T / sqrt(64)
// 8 waves/block, each wave: one 16-col N tile, 3 M tiles. Direct global frags.
// ---------------------------------------------------------------------------
__global__ __launch_bounds__(256)
void scores_kernel(const unsigned short* __restrict__ qred,
                   const unsigned short* __restrict__ kh,
                   float* __restrict__ scores) {
  const int bh = blockIdx.y;
  const int colBase = blockIdx.x * 128 + (threadIdx.x >> 5) * 16;
  const int lane = threadIdx.x & 31, r16 = lane & 15, hi = lane >> 4;
  const unsigned short* Ab = qred + (size_t)bh * UP * HD;
  const unsigned short* Bb = kh   + (size_t)bh * LL * HD;

  v8f acc[3] = {};
#pragma unroll
  for (int ks = 0; ks < HD; ks += 32) {
    FragAB bfr;
    const unsigned short* pb = Bb + (size_t)(colBase + r16) * HD + ks + hi * 16;
    bfr.q[0] = *(const uint4*)(pb - hi * 8);
    bfr.q[1] = *(const uint4*)(pb + 16 - hi * 8);
#pragma unroll
    for (int i = 0; i < 3; ++i) {
      FragAB af;
      const unsigned short* pa = Ab + (size_t)(i * 16 + r16) * HD + ks;
      af.q[0] = *(const uint4*)(pa + hi * 8);
      af.q[1] = *(const uint4*)(pa + 16 + hi * 8);
      acc[i] = __builtin_amdgcn_wmma_f32_16x16x32_bf16(
          false, af.v, false, bfr.v, (short)0, acc[i], false, false);
    }
  }
#pragma unroll
  for (int i = 0; i < 3; ++i)
#pragma unroll
    for (int r = 0; r < 8; ++r) {
      int row = i * 16 + hi * 8 + r;
      scores[((size_t)bh * UP + row) * LL + colBase + r16] = acc[i][r] * 0.125f;
    }
}

// row softmax over L, writes bf16 probs; padded rows -> 0
__global__ __launch_bounds__(256)
void softmax_kernel(const float* __restrict__ scores,
                    unsigned short* __restrict__ probs) {
  int t = blockIdx.x, bh = blockIdx.y, tid = threadIdx.x;
  const float* srow = scores + ((size_t)bh * UP + t) * LL;
  unsigned short* prow = probs + ((size_t)bh * UP + t) * LL;
  if (t >= UU) { for (int i = tid; i < LL; i += 256) prow[i] = 0; return; }
  __shared__ float red[256];
  float mx = -INFINITY;
  for (int i = tid; i < LL; i += 256) mx = fmaxf(mx, srow[i]);
  red[tid] = mx; __syncthreads();
  for (int s = 128; s > 0; s >>= 1) {
    if (tid < s) red[tid] = fmaxf(red[tid], red[tid + s]);
    __syncthreads();
  }
  float rmax = red[0]; __syncthreads();
  float sum = 0.0f;
  for (int i = tid; i < LL; i += 256) sum += __expf(srow[i] - rmax);
  red[tid] = sum; __syncthreads();
  for (int s = 128; s > 0; s >>= 1) {
    if (tid < s) red[tid] += red[tid + s];
    __syncthreads();
  }
  float inv = 1.0f / red[0];
  for (int i = tid; i < LL; i += 256)
    prow[i] = f32_bf16(__expf(srow[i] - rmax) * inv);
}

// ---------------------------------------------------------------------------
// outred(bh, UP, 64) = probs(UP, L) @ v(L, 64); B^T = vT(64, L). 4 waves.
// ---------------------------------------------------------------------------
__global__ __launch_bounds__(128)
void outred_kernel(const unsigned short* __restrict__ probs,
                   const unsigned short* __restrict__ vT,
                   float* __restrict__ outred) {
  const int bh = blockIdx.y, mt = blockIdx.x;       // mt in [0,3)
  const int wave = threadIdx.x >> 5;                // N tile
  const int lane = threadIdx.x & 31, r16 = lane & 15, hi = lane >> 4;
  const unsigned short* Ab = probs + ((size_t)bh * UP + mt * 16) * LL;
  const unsigned short* Bb = vT + ((size_t)bh * HD + wave * 16) * LL;

  v8f acc = {};
  for (int ks = 0; ks < LL; ks += 32) {
    FragAB af, bfr;
    const unsigned short* pa = Ab + (size_t)r16 * LL + ks;
    af.q[0] = *(const uint4*)(pa + hi * 8);
    af.q[1] = *(const uint4*)(pa + 16 + hi * 8);
    const unsigned short* pb = Bb + (size_t)r16 * LL + ks + hi * 16;
    bfr.q[0] = *(const uint4*)(pb - hi * 8);
    bfr.q[1] = *(const uint4*)(pb + 16 - hi * 8);
    acc = __builtin_amdgcn_wmma_f32_16x16x32_bf16(
        false, af.v, false, bfr.v, (short)0, acc, false, false);
  }
#pragma unroll
  for (int r = 0; r < 8; ++r) {
    int row = mt * 16 + hi * 8 + r;
    outred[((size_t)bh * UP + row) * HD + wave * 16 + r16] = acc[r];
  }
}

// ctx[(b,l,h*64+dd)] = outred[bh][t][dd] at l = Mtop[bh][t]
__global__ void scatter_ctx_kernel(const float* __restrict__ outred,
                                   const int* __restrict__ Mtop,
                                   unsigned short* __restrict__ ctx) {
  int i = blockIdx.x * 256 + threadIdx.x;     // BHN*UU*HD
  int dd2 = i & (HD - 1);
  int qq  = i >> 6;
  int t = qq % UU, bh = qq / UU;
  int b = bh >> 4, h = bh & 15;
  int l = Mtop[bh * UU + t];
  float v = outred[((size_t)bh * UP + t) * HD + dd2];
  ctx[((size_t)(b * LL) + l) * DD + h * HD + dd2] = f32_bf16(v);
}

// ---------------------------------------------------------------------------
// out = LayerNorm(xa + xb) * gamma + beta; optional bf16 mirror
// ---------------------------------------------------------------------------
__global__ __launch_bounds__(256)
void add_ln_kernel(const float* __restrict__ xa, const float* __restrict__ xb,
                   const float* __restrict__ gamma, const float* __restrict__ beta,
                   float* __restrict__ outf, unsigned short* __restrict__ outh) {
  __shared__ float red[256];
  int row = blockIdx.x, tid = threadIdx.x;
  const float* ar = xa + (size_t)row * DD;
  const float* br = xb + (size_t)row * DD;
  float vals[DD / 256];
  float s = 0.0f;
#pragma unroll
  for (int i = 0; i < DD / 256; ++i) {
    int c = tid + i * 256;
    vals[i] = ar[c] + br[c];
    s += vals[i];
  }
  red[tid] = s; __syncthreads();
  for (int st = 128; st > 0; st >>= 1) {
    if (tid < st) red[tid] += red[tid + st];
    __syncthreads();
  }
  float mean = red[0] * (1.0f / DD); __syncthreads();
  float vs = 0.0f;
#pragma unroll
  for (int i = 0; i < DD / 256; ++i) { float d = vals[i] - mean; vs += d * d; }
  red[tid] = vs; __syncthreads();
  for (int st = 128; st > 0; st >>= 1) {
    if (tid < st) red[tid] += red[tid + st];
    __syncthreads();
  }
  float inv = rsqrtf(red[0] * (1.0f / DD) + 1e-12f);
#pragma unroll
  for (int i = 0; i < DD / 256; ++i) {
    int c = tid + i * 256;
    float o = gamma[c] * (vals[i] - mean) * inv + beta[c];
    outf[(size_t)row * DD + c] = o;
    if (outh) outh[(size_t)row * DD + c] = f32_bf16(o);
  }
}

// ---------------------------------------------------------------------------
// Host-side orchestration
// ---------------------------------------------------------------------------
extern "C" void kernel_launch(void* const* d_in, const int* in_sizes, int n_in,
                              void* d_out, int out_size, void* d_ws, size_t ws_size,
                              hipStream_t stream) {
  (void)in_sizes; (void)n_in; (void)out_size; (void)ws_size;
  const float* x   = (const float*)d_in[0];
  const float* wq  = (const float*)d_in[1];
  const float* bq  = (const float*)d_in[2];
  const float* wk  = (const float*)d_in[3];
  const float* bk_ = (const float*)d_in[4];
  const float* wv  = (const float*)d_in[5];
  const float* bv  = (const float*)d_in[6];
  const float* wo  = (const float*)d_in[7];
  const float* bo  = (const float*)d_in[8];
  const float* w1  = (const float*)d_in[9];
  const float* b1  = (const float*)d_in[10];
  const float* w2  = (const float*)d_in[11];
  const float* b2  = (const float*)d_in[12];
  const float* g1  = (const float*)d_in[13];
  const float* be1 = (const float*)d_in[14];
  const float* g2  = (const float*)d_in[15];
  const float* be2 = (const float*)d_in[16];
  const int* isamp = (const int*)d_in[17];
  float* outp = (float*)d_out;

  char* ws = (char*)d_ws; size_t off = 0;
  auto alloc = [&](size_t bytes) -> void* {
    off = (off + 255) & ~(size_t)255;
    void* p = ws + off; off += bytes; return p;
  };

  unsigned short* xb    = (unsigned short*)alloc((size_t)NT * DD * 2);
  unsigned short* wqT   = (unsigned short*)alloc((size_t)DD * DD * 2);
  unsigned short* wkT   = (unsigned short*)alloc((size_t)DD * DD * 2);
  unsigned short* wvT   = (unsigned short*)alloc((size_t)DD * DD * 2);
  unsigned short* woT   = (unsigned short*)alloc((size_t)DD * DD * 2);
  unsigned short* w1T   = (unsigned short*)alloc((size_t)FF * DD * 2);
  unsigned short* w2T   = (unsigned short*)alloc((size_t)DD * FF * 2);
  unsigned short* qh    = (unsigned short*)alloc((size_t)BHN * LL * HD * 2);
  unsigned short* kh    = (unsigned short*)alloc((size_t)BHN * LL * HD * 2);
  unsigned short* vT    = (unsigned short*)alloc((size_t)BHN * HD * LL * 2);
  unsigned short* ksamp = (unsigned short*)alloc((size_t)BHN * USN * HD * 2);
  float*          Mmax  = (float*)alloc((size_t)BHN * LL * 4);
  float*          Msum  = (float*)alloc((size_t)BHN * LL * 4);
  int*            Mtop  = (int*)alloc((size_t)BHN * UU * 4);
  unsigned short* qred  = (unsigned short*)alloc((size_t)BHN * UP * HD * 2);
  float*          scrs  = (float*)alloc((size_t)BHN * UP * LL * 4);
  unsigned short* probs = (unsigned short*)alloc((size_t)BHN * UP * LL * 2);
  float*          ored  = (float*)alloc((size_t)BHN * UP * HD * 4);
  unsigned short* ctx   = (unsigned short*)alloc((size_t)NT * DD * 2);
  float*          abuf  = (float*)alloc((size_t)NT * DD * 4);   // attn out / ffn out
  float*          x1    = (float*)alloc((size_t)NT * DD * 4);
  unsigned short* x1b   = (unsigned short*)alloc((size_t)NT * DD * 2);
  unsigned short* h1b   = (unsigned short*)alloc((size_t)NT * FF * 2);

  // 1) converts + weight transposes (bf16)
  cvt_bf16_kernel<<<NT * DD / 256, 256, 0, stream>>>(x, xb, NT * DD);
  cvt_bf16_t_kernel<<<dim3(DD / 32, DD / 32), 256, 0, stream>>>(wq, wqT, DD, DD);
  cvt_bf16_t_kernel<<<dim3(DD / 32, DD / 32), 256, 0, stream>>>(wk, wkT, DD, DD);
  cvt_bf16_t_kernel<<<dim3(DD / 32, DD / 32), 256, 0, stream>>>(wv, wvT, DD, DD);
  cvt_bf16_t_kernel<<<dim3(DD / 32, DD / 32), 256, 0, stream>>>(wo, woT, DD, DD);
  cvt_bf16_t_kernel<<<dim3(FF / 32, DD / 32), 256, 0, stream>>>(w1, w1T, DD, FF);
  cvt_bf16_t_kernel<<<dim3(DD / 32, FF / 32), 256, 0, stream>>>(w2, w2T, FF, DD);

  // 2) projections: q is a flat reshape (plain layout), k/v head-major
  gemm_bf16_kernel<<<dim3(DD / GM_BN, NT / GM_BM), 256, 0, stream>>>(
      xb, DD, wqT, DD, bq, DD, DD, MODE_PLAIN, qh);
  gemm_bf16_kernel<<<dim3(DD / GM_BN, NT / GM_BM), 256, 0, stream>>>(
      xb, DD, wkT, DD, bk_, DD, DD, MODE_HEADS, kh);
  gemm_bf16_kernel<<<dim3(DD / GM_BN, NT / GM_BM), 256, 0, stream>>>(
      xb, DD, wvT, DD, bv, DD, DD, MODE_VT, vT);

  // 3) sampled-score pass + M statistics + top-k query selection
  gather_ksamp_kernel<<<BHN * USN * HD / 256, 256, 0, stream>>>(kh, isamp, ksamp);
  init_m_kernel<<<BHN * LL / 256, 256, 0, stream>>>(Mmax, Msum);
  qk_m_kernel<<<dim3(USN / 64, LL / 128, BHN), 256, 0, stream>>>(qh, ksamp, Mmax, Msum);
  topk_kernel<<<BHN, 256, 0, stream>>>(Mmax, Msum, Mtop);

  // 4) sparse attention on selected queries
  gather_qred_kernel<<<BHN * UP * HD / 256, 256, 0, stream>>>(qh, Mtop, qred);
  scores_kernel<<<dim3(LL / 128, BHN), 256, 0, stream>>>(qred, kh, scrs);
  softmax_kernel<<<dim3(UP, BHN), 256, 0, stream>>>(scrs, probs);
  outred_kernel<<<dim3(UP / 16, BHN), 128, 0, stream>>>(probs, vT, ored);

  // 5) scatter into context, output projection, residual + LN1
  zero_u16_kernel<<<NT * DD / 256, 256, 0, stream>>>(ctx, NT * DD);
  scatter_ctx_kernel<<<BHN * UU * HD / 256, 256, 0, stream>>>(ored, Mtop, ctx);
  gemm_bf16_kernel<<<dim3(DD / GM_BN, NT / GM_BM), 256, 0, stream>>>(
      ctx, DD, woT, DD, bo, DD, DD, MODE_F32, abuf);
  add_ln_kernel<<<NT, 256, 0, stream>>>(x, abuf, g1, be1, x1, x1b);

  // 6) FFN + residual + LN2
  gemm_bf16_kernel<<<dim3(FF / GM_BN, NT / GM_BM), 256, 0, stream>>>(
      x1b, DD, w1T, DD, b1, FF, DD, MODE_RELU, h1b);
  gemm_bf16_kernel<<<dim3(DD / GM_BN, NT / GM_BM), 256, 0, stream>>>(
      h1b, FF, w2T, FF, b2, DD, FF, MODE_F32, abuf);
  add_ln_kernel<<<NT, 256, 0, stream>>>(x1, abuf, g2, be2, outp, nullptr);
}
